// Net_74363063763034
// MI455X (gfx1250) — matrix operand order repaired and checked by reading, not compiled, not run on hardware
//
#include <hip/hip_runtime.h>
#include <hip/hip_bf16.h>

typedef __attribute__((ext_vector_type(16))) _Float16 v16h;
typedef __attribute__((ext_vector_type(8)))  _Float16 v8h;
typedef __attribute__((ext_vector_type(8)))  float    v8f;

union frag16 { v16h v; v8h h[2]; };

#define TPB 256

// ---------------------------------------------------------------------------
// Generic direct convolution (small strided encoder layers and 1x1 convs).
// ---------------------------------------------------------------------------
template <int CIN, int COUT, int KS, int S, int P, bool RELU, bool IN16, bool OUT16>
__global__ __launch_bounds__(TPB) void conv_direct(
    const void* __restrict__ in_, const float* __restrict__ w,
    const float* __restrict__ bias, void* __restrict__ out_,
    int N, int Hin, int Win, int Hout, int Wout,
    int in_ct, int in_co, int out_ct, int out_co) {
  int idx = blockIdx.x * TPB + threadIdx.x;
  int total = N * COUT * Hout * Wout;
  if (idx >= total) return;
  int x = idx % Wout; int t = idx / Wout;
  int y = t % Hout;   t /= Hout;
  int co = t % COUT;  int n = t / COUT;

  const float*    inf = (const float*)in_;
  const _Float16* inh = (const _Float16*)in_;

  float acc = bias[co];
  for (int ci = 0; ci < CIN; ++ci) {
    size_t ibase = (size_t)(n * in_ct + in_co + ci) * Hin;
    const float* wp = w + ((co * CIN + ci) * KS) * KS;
#pragma unroll
    for (int ky = 0; ky < KS; ++ky) {
      int yi = y * S + ky - P;
      if (yi < 0 || yi >= Hin) continue;
#pragma unroll
      for (int kx = 0; kx < KS; ++kx) {
        int xi = x * S + kx - P;
        if (xi < 0 || xi >= Win) continue;
        size_t ii = (ibase + yi) * (size_t)Win + xi;
        float v = IN16 ? (float)inh[ii] : inf[ii];
        acc += v * wp[ky * KS + kx];
      }
    }
  }
  if (RELU) acc = fmaxf(acc, 0.f);
  size_t oidx = ((size_t)(n * out_ct + out_co + co) * Hout + y) * Wout + x;
  if (OUT16) ((_Float16*)out_)[oidx] = (_Float16)acc;
  else       ((float*)out_)[oidx]    = acc;
}

// ---------------------------------------------------------------------------
// WMMA implicit-GEMM 3x3 s1 p1 conv at 64x64 (NCHW), f16 in, f32 accumulate.
// Block = 128 threads (4 wave32s) = one output row of one image; wave w owns
// pixels x = 16w..16w+15; blockIdx.y tiles Cout by 16.
//
// K is ordered tap-major: k = tap*CINP + cin (CINP = Cin padded to 16), and
// LDS layouts are chosen so every A/B fragment half-run (8 consecutive K per
// lane) is 8 contiguous f16 -> each fragment is exactly two ds_load_b128:
//   activations: [4 rows][68 cols][CSTR]  (cin innermost, padded stride)
//   weights:     [16 n][KPAD k]
// K padded to a multiple of 32 (zeroed tail row / weights) so the WMMA loop
// is divergence free (EXEC must stay all ones).
// ---------------------------------------------------------------------------
template <int CIN, int COUT, bool RELU, bool OUT16>
__global__ __launch_bounds__(128) void conv3x3_wmma(
    const _Float16* __restrict__ in, const float* __restrict__ w,
    const float* __restrict__ bias, float* __restrict__ out_f32,
    _Float16* __restrict__ out_f16) {
  constexpr int CINP = ((CIN + 15) / 16) * 16;    // 16-aligned cin
  constexpr int CSTR = CINP + 8;                  // LDS stride (bank stagger)
  constexpr int KPAD = ((9 * CINP + 31) / 32) * 32;

  __shared__ __align__(16) _Float16 actl[4 * 68 * CSTR]; // [row][col][cin]
  __shared__ __align__(16) _Float16 wl[16 * KPAD];       // [n][k]

  const int tid  = threadIdx.x;
  const int lane = tid & 31;
  const int wv   = tid >> 5;
  const int img  = blockIdx.x >> 6;
  const int y    = blockIdx.x & 63;
  const int ntile = blockIdx.y;

  // Stage activations: read coalesced along cols, write cin-innermost.
  // Covers rows 0..3 (row 3 = zero pad for the K tail) and halo cols.
  for (int t = tid; t < CINP * 272; t += 128) {
    int cin = t / 272; int rr = t % 272;
    int r = rr / 68;   int col = rr % 68;
    int yin = y + r - 1;
    _Float16 v = (_Float16)0.f;
    if (r < 3 && cin < CIN && yin >= 0 && yin < 64 && col >= 1 && col <= 64)
      v = in[((size_t)(img * CIN + cin) * 64 + yin) * 64 + (col - 1)];
    actl[(r * 68 + col) * CSTR + cin] = v;
  }
  // Stage weights f32->f16, tap-major K, zero padded in cin/tap/n.
  for (int t = tid; t < 16 * KPAD; t += 128) {
    int k = t % KPAD;
    int ng = ntile * 16 + t / KPAD;
    int tap = k / CINP; int cin = k - tap * CINP;
    float v = 0.f;
    if (tap < 9 && cin < CIN && ng < COUT)
      v = w[(size_t)ng * CIN * 9 + cin * 9 + tap];
    wl[t] = (_Float16)v;
  }
  __syncthreads();

  const int xbase = wv * 16 + (lane & 15);   // this lane's M row / pixel x
  const int khalf = (lane < 16) ? 0 : 8;     // A layout half-wave K offset
  const int nb    = lane & 15;               // B column
  const int kbadd = (lane < 16) ? 0 : 8;     // B layout half-wave K offset

  v8f acc = {};
#pragma unroll 1
  for (int kc = 0; kc < KPAD; kc += 32) {
    // A fragment: elements 0..7 = K kc+khalf.., 8..15 = K kc+16+khalf..
    int k0 = kc + khalf;
    int k1 = k0 + 16;
    int t0 = k0 / CINP, c0 = k0 - t0 * CINP;
    int t1 = k1 / CINP, c1 = k1 - t1 * CINP;
    int dy0 = t0 / 3, dx0 = t0 - 3 * dy0;    // tail tap 9 -> row 3 (zeros)
    int dy1 = t1 / 3, dx1 = t1 - 3 * dy1;
    frag16 a, b;
    a.h[0] = *(const v8h*)&actl[(dy0 * 68 + xbase + dx0) * CSTR + c0];
    a.h[1] = *(const v8h*)&actl[(dy1 * 68 + xbase + dx1) * CSTR + c1];
    // B fragment: elements 0..7 = K kc+kbadd.., 8..15 = K kc+16+kbadd.. at column nb.
    const _Float16* bp = &wl[nb * KPAD + kc + kbadd];
    b.h[0] = *(const v8h*)bp;
    b.h[1] = *(const v8h*)(bp + 16);
    acc = __builtin_amdgcn_wmma_f32_16x16x32_f16(
        false, a.v, false, b.v, (short)0, acc, false, false);
  }

  // D layout: column N = lane&15; element r holds M = (lane<16?0:8)+r.
  const int ng = ntile * 16 + (lane & 15);
  if (ng < COUT) {
    const int m0 = (lane < 16) ? 0 : 8;
    const float bz = bias[ng];
    size_t base = ((size_t)(img * COUT + ng) * 64 + y) * 64 + wv * 16 + m0;
    if (OUT16) {
      v8h h;
#pragma unroll
      for (int r = 0; r < 8; ++r) {
        float v = acc[r] + bz;
        if (RELU) v = fmaxf(v, 0.f);
        h[r] = (_Float16)v;
      }
      *(v8h*)(out_f16 + base) = h;             // 16B aligned vector store
    } else {
      float4 v0, v1;
      v0.x = acc[0] + bz; v0.y = acc[1] + bz; v0.z = acc[2] + bz; v0.w = acc[3] + bz;
      v1.x = acc[4] + bz; v1.y = acc[5] + bz; v1.z = acc[6] + bz; v1.w = acc[7] + bz;
      *(float4*)(out_f32 + base)     = v0;
      *(float4*)(out_f32 + base + 4) = v1;
    }
  }
}

// ---------------------------------------------------------------------------
// Broadcast the 16 bottleneck scalars per image into 64x64 blocks
// (channels 0..15 of the 48-channel concat buffer, f16).
// ---------------------------------------------------------------------------
__global__ __launch_bounds__(TPB) void blur_fill(
    const float* __restrict__ a7, _Float16* __restrict__ pre, int total) {
  int idx = blockIdx.x * TPB + threadIdx.x;
  if (idx >= total) return;
  int sp = idx & 4095; int c = (idx >> 12) & 15; int n = idx >> 16;
  pre[((size_t)(n * 48 + c) << 12) + sp] = (_Float16)a7[n * 16 + c];
}

// ---------------------------------------------------------------------------
// BatchNorm (training mode, biased batch stats over N,H,W at 64x64, N=128).
// ---------------------------------------------------------------------------
__global__ __launch_bounds__(TPB) void zero_f32(float* p, int n) {
  int i = blockIdx.x * TPB + threadIdx.x;
  if (i < n) p[i] = 0.f;
}

__global__ __launch_bounds__(TPB) void bn_reduce(
    const float* __restrict__ x, float* __restrict__ sum,
    float* __restrict__ sumsq, int C) {
  __shared__ float s1[TPB], s2[TPB];
  int plane = blockIdx.x;                 // n*C + c  (matches memory layout)
  int c = plane % C;
  const float* p = x + (size_t)plane * 4096;
  float a = 0.f, b = 0.f;
  for (int i = threadIdx.x; i < 4096; i += TPB) {
    float v = p[i]; a += v; b += v * v;
  }
  s1[threadIdx.x] = a; s2[threadIdx.x] = b;
  __syncthreads();
  for (int st = TPB / 2; st > 0; st >>= 1) {
    if (threadIdx.x < st) {
      s1[threadIdx.x] += s1[threadIdx.x + st];
      s2[threadIdx.x] += s2[threadIdx.x + st];
    }
    __syncthreads();
  }
  if (threadIdx.x == 0) {
    atomicAdd(&sum[c], s1[0]);
    atomicAdd(&sumsq[c], s2[0]);
  }
}

template <bool ADD, bool RELU>
__global__ __launch_bounds__(TPB) void bn_apply(
    const float* __restrict__ x, const float* __restrict__ g,
    const float* __restrict__ be, const float* __restrict__ sum,
    const float* __restrict__ sumsq, const float* __restrict__ add,
    _Float16* __restrict__ out, int C, int total) {
  int idx = blockIdx.x * TPB + threadIdx.x;
  if (idx >= total) return;
  int c = (idx >> 12) % C;
  const float invM = 1.f / 524288.f;      // N*H*W = 128*64*64
  float mu  = sum[c] * invM;
  float var = sumsq[c] * invM - mu * mu;
  float v = g[c] * (x[idx] - mu) * rsqrtf(var + 1e-5f) + be[c];
  if (ADD)  v += add[idx];
  if (RELU) v = fmaxf(v, 0.f);
  out[idx] = (_Float16)v;
}

// ---------------------------------------------------------------------------
// Fused u2 (1x1, 30->16) + pixel_shuffle(r=4) + sigmoid -> [N,1,256,256] f32.
// ---------------------------------------------------------------------------
__global__ __launch_bounds__(TPB) void final_shuffle(
    const _Float16* __restrict__ u1, const float* __restrict__ w2,
    const float* __restrict__ b2, float* __restrict__ out, int total) {
  int idx = blockIdx.x * TPB + threadIdx.x;
  if (idx >= total) return;
  int W = idx & 255; int H = (idx >> 8) & 255; int n = idx >> 16;
  int c = (H & 3) * 4 + (W & 3);
  int h = H >> 2;   int ww = W >> 2;
  float acc = b2[c];
  const _Float16* pin = u1 + (size_t)n * 30 * 4096 + h * 64 + ww;
#pragma unroll
  for (int ci = 0; ci < 30; ++ci)
    acc += (float)pin[(size_t)ci * 4096] * w2[c * 30 + ci];
  acc = fmaxf(acc, 0.f);
  out[idx] = 1.f / (1.f + expf(-acc));
}

// ---------------------------------------------------------------------------
extern "C" void kernel_launch(void* const* d_in, const int* in_sizes, int n_in,
                              void* d_out, int out_size, void* d_ws, size_t ws_size,
                              hipStream_t stream) {
  (void)in_sizes; (void)n_in; (void)out_size; (void)ws_size;
  const int N = 128;

  const float* x   = (const float*)d_in[0];
  const float* c1w = (const float*)d_in[1];  const float* c1b = (const float*)d_in[2];
  const float* c2w = (const float*)d_in[3];  const float* c2b = (const float*)d_in[4];
  const float* c3w = (const float*)d_in[5];  const float* c3b = (const float*)d_in[6];
  const float* c4w = (const float*)d_in[7];  const float* c4b = (const float*)d_in[8];
  const float* c5w = (const float*)d_in[9];  const float* c5b = (const float*)d_in[10];
  const float* c6w = (const float*)d_in[11]; const float* c6b = (const float*)d_in[12];
  const float* c7w = (const float*)d_in[13]; const float* c7b = (const float*)d_in[14];
  const float* r1w = (const float*)d_in[15]; const float* r1b = (const float*)d_in[16];
  const float* g1  = (const float*)d_in[17]; const float* be1 = (const float*)d_in[18];
  const float* r2w = (const float*)d_in[19]; const float* r2b = (const float*)d_in[20];
  const float* g2  = (const float*)d_in[21]; const float* be2 = (const float*)d_in[22];
  const float* pww = (const float*)d_in[23]; const float* pwb = (const float*)d_in[24];
  const float* u1w = (const float*)d_in[25]; const float* u1b = (const float*)d_in[26];
  const float* u2w = (const float*)d_in[27]; const float* u2b = (const float*)d_in[28];

  // -------- workspace arena (offsets 256B aligned); r1f32 aliases dead a1 --
  char* ws = (char*)d_ws;
  size_t off = 0;
  auto arena = [&](size_t bytes) {
    void* p = ws + off;
    off += (bytes + 255) & ~(size_t)255;
    return p;
  };
  float*    r1f32 = (float*)arena((size_t)33554432 * 4); // [128,64,64,64]
  float*    a1    = r1f32;                               // [128,8,128,128] (disjoint lifetime)
  _Float16* pre16 = (_Float16*)arena((size_t)25165824 * 2); // [128,48,64,64]
  float*    a3    = (float*)arena((size_t)524288 * 4);   // [128,4,32,32]
  float*    a4    = (float*)arena((size_t)655360 * 4);   // [128,20,16,16]
  float*    a5    = (float*)arena((size_t)245760 * 4);   // [128,30,8,8]
  float*    a6    = (float*)arena((size_t)16384 * 4);    // [128,8,4,4]
  float*    a7    = (float*)arena((size_t)2048 * 4);     // [128,16]
  _Float16* r1f16 = (_Float16*)arena((size_t)33554432 * 2);
  float*    r2f32 = (float*)arena((size_t)10485760 * 4); // [128,20,64,64]
  float*    pwout = (float*)arena((size_t)10485760 * 4);
  _Float16* r2y16 = (_Float16*)arena((size_t)10485760 * 2);
  _Float16* u1f16 = (_Float16*)arena((size_t)15728640 * 2); // [128,30,64,64]
  float*    stats = (float*)arena((size_t)256 * 4);
  float* sum1 = stats;       float* ssq1 = stats + 64;
  float* sum2 = stats + 128; float* ssq2 = stats + 148;

  auto blks = [](int total) { return (total + TPB - 1) / TPB; };

  // -------- encoder (direct convs) ----------------------------------------
  conv_direct<1, 8, 3, 2, 1, true, false, false><<<blks(16777216), TPB, 0, stream>>>(
      x, c1w, c1b, a1, N, 256, 256, 128, 128, 1, 0, 8, 0);
  // c2 writes f16 directly into channels 16..47 of the concat buffer
  conv_direct<8, 32, 5, 2, 2, true, false, true><<<blks(16777216), TPB, 0, stream>>>(
      a1, c2w, c2b, pre16, N, 128, 128, 64, 64, 8, 0, 48, 16);
  conv_direct<32, 4, 3, 2, 1, false, true, false><<<blks(524288), TPB, 0, stream>>>(
      pre16, c3w, c3b, a3, N, 64, 64, 32, 32, 48, 16, 4, 0);
  conv_direct<4, 20, 3, 2, 1, true, false, false><<<blks(655360), TPB, 0, stream>>>(
      a3, c4w, c4b, a4, N, 32, 32, 16, 16, 4, 0, 20, 0);
  conv_direct<20, 30, 5, 2, 2, true, false, false><<<blks(245760), TPB, 0, stream>>>(
      a4, c5w, c5b, a5, N, 16, 16, 8, 8, 20, 0, 30, 0);
  conv_direct<30, 8, 3, 2, 1, true, false, false><<<blks(16384), TPB, 0, stream>>>(
      a5, c6w, c6b, a6, N, 8, 8, 4, 4, 30, 0, 8, 0);
  conv_direct<8, 1, 1, 1, 0, true, false, false><<<blks(2048), TPB, 0, stream>>>(
      a6, c7w, c7b, a7, N, 4, 4, 4, 4, 8, 0, 1, 0);

  // bottleneck broadcast -> channels 0..15 of concat buffer
  blur_fill<<<blks(8388608), TPB, 0, stream>>>(a7, pre16, 8388608);

  // 1x1 skip projection (48->20) on the concat buffer
  conv_direct<48, 20, 1, 1, 0, false, true, false><<<blks(10485760), TPB, 0, stream>>>(
      pre16, pww, pwb, pwout, N, 64, 64, 64, 64, 48, 0, 20, 0);

  zero_f32<<<1, TPB, 0, stream>>>(stats, 256);

  // -------- r1: WMMA conv 48->64, then BN(batch stats)+ReLU ---------------
  conv3x3_wmma<48, 64, false, false><<<dim3(N * 64, 4), 128, 0, stream>>>(
      pre16, r1w, r1b, r1f32, (_Float16*)nullptr);
  bn_reduce<<<N * 64, TPB, 0, stream>>>(r1f32, sum1, ssq1, 64);
  bn_apply<false, true><<<blks(33554432), TPB, 0, stream>>>(
      r1f32, g1, be1, sum1, ssq1, (const float*)nullptr, r1f16, 64, 33554432);

  // -------- r2: WMMA conv 64->20, BN + skip add ---------------------------
  conv3x3_wmma<64, 20, false, false><<<dim3(N * 64, 2), 128, 0, stream>>>(
      r1f16, r2w, r2b, r2f32, (_Float16*)nullptr);
  bn_reduce<<<N * 20, TPB, 0, stream>>>(r2f32, sum2, ssq2, 20);
  bn_apply<true, false><<<blks(10485760), TPB, 0, stream>>>(
      r2f32, g2, be2, sum2, ssq2, pwout, r2y16, 20, 10485760);

  // -------- u1: WMMA conv 20->30 + ReLU, f16 out --------------------------
  conv3x3_wmma<20, 30, true, true><<<dim3(N * 64, 2), 128, 0, stream>>>(
      r2y16, u1w, u1b, (float*)nullptr, u1f16);

  // -------- u2 (1x1 30->16) + pixel_shuffle(4) + sigmoid ------------------
  final_shuffle<<<blks(8388608), TPB, 0, stream>>>(
      u1f16, u2w, u2b, (float*)d_out, 8388608);
}
